// RNN_2525440770543
// MI455X (gfx1250) — matrix operand order repaired and checked by reading, not compiled, not run on hardware
//
#include <hip/hip_runtime.h>
#include <hip/hip_bf16.h>

typedef __attribute__((ext_vector_type(16))) _Float16 v16h;
typedef __attribute__((ext_vector_type(8)))  float    v8f;

#define B_   2
#define T_   10
#define H_   96
#define W_   96
#define HW_  (H_ * W_)
#define NH_  64
#define IMGC 10
#define ACTC 7
#define STATC 12

// ---------------------------------------------------------------------------
// Generic conv (KxK, pad=K/2) as implicit GEMM on the WMMA pipe.
// One wave per block computes a 32(Cout) x 32(pixels) tile; K marched in
// 32-chunks; per chunk: 2 A-frags x 2 B-frags -> 4 v_wmma_f32_16x16x32_f16.
// f16 operands staged in LDS, f32 accumulation.
// ---------------------------------------------------------------------------
__global__ __launch_bounds__(32)
void conv_wmma_kernel(const float* __restrict__ in, long in_bstride,
                      const float* __restrict__ w, const float* __restrict__ bias,
                      float* __restrict__ out,
                      int Cin, int Cout, int Ksz)
{
    const int pad  = Ksz >> 1;
    const int Ktot = Cin * Ksz * Ksz;
    const int wtiles = W_ / 32;           // 3

    int mt = blockIdx.x;
    int pt = blockIdx.y;
    int b  = pt / (H_ * wtiles);
    int r0 = pt % (H_ * wtiles);
    int y  = r0 / wtiles;
    int x0 = (r0 % wtiles) * 32;
    int m0 = mt * 32;

    __shared__ _Float16 As[32 * 32];      // 32 Cout rows x 32 K
    __shared__ _Float16 Bs[32 * 32];      // 32 K rows x 32 pixels

    int lane = threadIdx.x;
    int row  = lane & 15;
    int half = lane >> 4;

    v8f acc00 = {}, acc01 = {}, acc10 = {}, acc11 = {};
    const float* inb = in + (long)b * in_bstride;

    for (int k0 = 0; k0 < Ktot; k0 += 32) {
        // --- stage A tile: lane t loads weight row (m0+t), 32 consecutive K ---
        {
            const float* wrow = w + (long)(m0 + lane) * Ktot;
            bool rok = (m0 + lane) < Cout;
            if (rok && (k0 + 32) < Ktot)
                __builtin_prefetch(wrow + k0 + 32, 0, 3);   // global_prefetch_b8
            #pragma unroll
            for (int i = 0; i < 32; ++i) {
                int k = k0 + i;
                As[lane * 32 + i] = (rok && k < Ktot) ? (_Float16)wrow[k]
                                                      : (_Float16)0.0f;
            }
        }
        // --- stage B tile: lane t is K-row t, im2col 32 pixels on the fly ---
        {
            int k = k0 + lane;
            _Float16 vals[32];
            if (k < Ktot) {
                int cin = k / (Ksz * Ksz);
                int kk  = k % (Ksz * Ksz);
                int ky  = kk / Ksz - pad;
                int kx  = kk % Ksz - pad;
                int iy  = y + ky;
                if (iy >= 0 && iy < H_) {
                    const float* ip = inb + ((long)cin * H_ + iy) * W_;
                    __builtin_prefetch(ip + W_, 0, 3);
                    #pragma unroll
                    for (int px = 0; px < 32; ++px) {
                        int ix = x0 + px + kx;
                        vals[px] = (ix >= 0 && ix < W_) ? (_Float16)ip[ix]
                                                        : (_Float16)0.0f;
                    }
                } else {
                    #pragma unroll
                    for (int px = 0; px < 32; ++px) vals[px] = (_Float16)0.0f;
                }
            } else {
                #pragma unroll
                for (int px = 0; px < 32; ++px) vals[px] = (_Float16)0.0f;
            }
            #pragma unroll
            for (int px = 0; px < 32; ++px) Bs[lane * 32 + px] = vals[px];
        }
        __syncthreads();

        // --- pack fragments per gfx1250 16-bit A(16x32)/B(32x16) layouts ---
        v16h a0, a1, bf0, bf1;
        #pragma unroll
        for (int e = 0; e < 16; ++e) {
            int ka = (e & 7) + ((e >> 3) << 4) + (half << 3);   // A: ISA table
            a0[e] = As[row * 32 + ka];
            a1[e] = As[(row + 16) * 32 + ka];
            int kb = (e & 7) + ((e >> 3) << 4) + (half << 3);   // B: ISA table
            bf0[e] = Bs[kb * 32 + row];
            bf1[e] = Bs[kb * 32 + row + 16];
        }
        acc00 = __builtin_amdgcn_wmma_f32_16x16x32_f16(false, a0, false, bf0,
                                                       (short)0, acc00, false, false);
        acc01 = __builtin_amdgcn_wmma_f32_16x16x32_f16(false, a0, false, bf1,
                                                       (short)0, acc01, false, false);
        acc10 = __builtin_amdgcn_wmma_f32_16x16x32_f16(false, a1, false, bf0,
                                                       (short)0, acc10, false, false);
        acc11 = __builtin_amdgcn_wmma_f32_16x16x32_f16(false, a1, false, bf1,
                                                       (short)0, acc11, false, false);
        __syncthreads();
    }

    // D layout: lane = N column, VGPR r -> M = half*8 + r
    #pragma unroll
    for (int r = 0; r < 8; ++r) {
        int m   = (half << 3) + r;
        int oc0 = m0 + m;
        int oc1 = m0 + 16 + m;
        if (oc0 < Cout) {
            float bv = bias ? bias[oc0] : 0.0f;
            float* op = out + (((long)b * Cout + oc0) * H_ + y) * W_ + x0;
            op[row]      = acc00[r] + bv;
            op[row + 16] = acc01[r] + bv;
        }
        if (oc1 < Cout) {
            float bv = bias ? bias[oc1] : 0.0f;
            float* op = out + (((long)b * Cout + oc1) * H_ + y) * W_ + x0;
            op[row]      = acc10[r] + bv;
            op[row + 16] = acc11[r] + bv;
        }
    }
}

// ---------------------------------------------------------------------------
__device__ __forceinline__ float sigmf(float x) { return 1.0f / (1.0f + __expf(-x)); }

__global__ void split_c_kernel(const float* __restrict__ call,
                               float* __restrict__ c0, float* __restrict__ c1)
{
    long i = (long)blockIdx.x * blockDim.x + threadIdx.x;
    long total = (long)B_ * 2 * NH_ * HW_;
    if (i >= total) return;
    int p  = i % HW_;
    int ch = (i / HW_) % (2 * NH_);
    int b  = i / ((long)2 * NH_ * HW_);
    float v = call[i];
    if (ch < NH_) c0[((long)b * NH_ + ch) * HW_ + p] = v;
    else          c1[((long)b * NH_ + ch - NH_) * HW_ + p] = v;
}

__global__ void cell_pointwise_kernel(const float* __restrict__ xc,
                                      const float* __restrict__ hc,
                                      const float* __restrict__ ac,
                                      const float* __restrict__ mc,
                                      float* __restrict__ cbuf,
                                      float* __restrict__ mbuf,
                                      float* __restrict__ memcat,
                                      float* __restrict__ dcb,
                                      float* __restrict__ dmb)
{
    long i = (long)blockIdx.x * blockDim.x + threadIdx.x;
    long total = (long)B_ * NH_ * HW_;
    if (i >= total) return;
    int p = i % HW_;
    int j = (i / HW_) % NH_;
    int b = i / ((long)NH_ * HW_);
    long step = (long)NH_ * HW_;
    long sx = (long)b * 7 * step + (long)j * HW_ + p;
    long s4 = (long)b * 4 * step + (long)j * HW_ + p;
    long s3 = (long)b * 3 * step + (long)j * HW_ + p;
    long s1 = ((long)b * NH_ + j) * HW_ + p;

    float hc0 = hc[s4]            * ac[s4];
    float hc1 = hc[s4 + step]     * ac[s4 + step];
    float hc2 = hc[s4 + 2 * step] * ac[s4 + 2 * step];

    float it = sigmf(xc[sx]            + hc0);
    float ft = sigmf(xc[sx + step]     + hc1 + 1.0f);
    float gt = tanhf(xc[sx + 2 * step] + hc2);
    float dc = it * gt;
    float cn = ft * cbuf[s1] + dc;

    float itp = sigmf(xc[sx + 3 * step] + mc[s3]);
    float ftp = sigmf(xc[sx + 4 * step] + mc[s3 + step] + 1.0f);
    float gtp = tanhf(xc[sx + 5 * step] + mc[s3 + 2 * step]);
    float dm = itp * gtp;
    float mn = ftp * mbuf[s1] + dm;

    cbuf[s1] = cn;  mbuf[s1] = mn;  dcb[s1] = dc;  dmb[s1] = dm;
    long mi = ((long)b * 2 * NH_ + j) * HW_ + p;
    memcat[mi] = cn;  memcat[mi + step] = mn;
}

__global__ void cell_output_kernel(const float* __restrict__ xc,
                                   const float* __restrict__ hc,
                                   const float* __restrict__ ac,
                                   const float* __restrict__ oconv,
                                   const float* __restrict__ wclc,
                                   float* __restrict__ hbuf)
{
    long i = (long)blockIdx.x * blockDim.x + threadIdx.x;
    long total = (long)B_ * NH_ * HW_;
    if (i >= total) return;
    int p = i % HW_;
    int j = (i / HW_) % NH_;
    int b = i / ((long)NH_ * HW_);
    long step = (long)NH_ * HW_;
    long s1 = ((long)b * NH_ + j) * HW_ + p;
    float o_x = xc[(long)b * 7 * step + 6 * step + (long)j * HW_ + p];
    long s4o  = (long)b * 4 * step + 3 * step + (long)j * HW_ + p;
    float o_h = hc[s4o] * ac[s4o];
    float ot = sigmf(o_x + o_h + oconv[s1]);
    hbuf[s1] = ot * tanhf(wclc[s1]);
}

__global__ __launch_bounds__(256)
void cos_reduce_kernel(const float* __restrict__ yc, const float* __restrict__ ym,
                       float* __restrict__ dec_acc, float scale)
{
    const float* a = yc + (long)blockIdx.x * HW_;
    const float* c = ym + (long)blockIdx.x * HW_;
    float sab = 0.f, saa = 0.f, sbb = 0.f;
    for (int p = threadIdx.x; p < HW_; p += 256) {
        float av = a[p], bv = c[p];
        sab += av * bv; saa += av * av; sbb += bv * bv;
    }
    __shared__ float r0[256], r1[256], r2[256];
    int tid = threadIdx.x;
    r0[tid] = sab; r1[tid] = saa; r2[tid] = sbb;
    __syncthreads();
    for (int s = 128; s > 0; s >>= 1) {
        if (tid < s) { r0[tid] += r0[tid + s]; r1[tid] += r1[tid + s]; r2[tid] += r2[tid + s]; }
        __syncthreads();
    }
    if (tid == 0) {
        float Na = fmaxf(sqrtf(r1[0]), 1e-12f);
        float Nb = fmaxf(sqrtf(r2[0]), 1e-12f);
        float dn = r0[0] / (Na * Nb);
        float n1 = sqrtf(r1[0]) / Na, n2 = sqrtf(r2[0]) / Nb;
        float val = fabsf(dn / fmaxf(n1 * n2, 1e-8f));
        atomicAdd(dec_acc, val * scale);
    }
}

__global__ void frame_kernel(const float* __restrict__ h, const float* __restrict__ w,
                             float* __restrict__ net, float* __restrict__ dout, int t)
{
    long i = (long)blockIdx.x * blockDim.x + threadIdx.x;
    long total = (long)B_ * IMGC * HW_;
    if (i >= total) return;
    int p = i % HW_;
    int c = (i / HW_) % IMGC;
    int b = i / ((long)IMGC * HW_);
    const float* hb = h + (long)b * NH_ * HW_ + p;
    float s = 0.f;
    #pragma unroll 8
    for (int k = 0; k < NH_; ++k) s += w[c * NH_ + k] * hb[(long)k * HW_];
    long ni = ((long)b * IMGC + c) * HW_ + p;
    float v = s + net[ni];
    net[ni] = v;
    dout[(((long)b * T_ + t) * IMGC + c) * HW_ + p] = v;
}

__global__ __launch_bounds__(256)
void mse_kernel(const float* __restrict__ pred, const float* __restrict__ tgt,
                float* __restrict__ acc, long n)
{
    float s = 0.f;
    for (long i = (long)blockIdx.x * blockDim.x + threadIdx.x; i < n;
         i += (long)gridDim.x * blockDim.x) {
        float d = pred[i] - tgt[i];
        s += d * d;
    }
    __shared__ float r[256];
    int tid = threadIdx.x;
    r[tid] = s;
    __syncthreads();
    for (int st = 128; st > 0; st >>= 1) {
        if (tid < st) r[tid] += r[tid + st];
        __syncthreads();
    }
    if (tid == 0) atomicAdd(acc, r[0]);
}

__global__ void finalize_kernel(const float* __restrict__ acc, float* __restrict__ loss, long n)
{
    loss[0] = acc[0] / (float)n + 0.1f * acc[1];
}

// ---------------------------------------------------------------------------
static inline void launch_conv(const float* in, long bstride, const float* w,
                               const float* bias, float* out,
                               int Cin, int Cout, int Ksz, hipStream_t s)
{
    dim3 g((Cout + 31) / 32, B_ * H_ * (W_ / 32));
    conv_wmma_kernel<<<g, 32, 0, s>>>(in, bstride, w, bias, out, Cin, Cout, Ksz);
}

extern "C" void kernel_launch(void* const* d_in, const int* in_sizes, int n_in,
                              void* d_out, int out_size, void* d_ws, size_t ws_size,
                              hipStream_t stream)
{
    // ---- identify inputs by flat element count (robust to flatten order) ----
    const float *forc = nullptr, *init = nullptr, *stat = nullptr, *tgt = nullptr;
    const float *wx[2] = {}, *wa[2] = {}, *wh[2] = {}, *wm[2] = {}, *wo[2] = {}, *wcl[2] = {};
    const float *adapter = nullptr, *conv_last = nullptr, *mem_w = nullptr,
                *mem_b = nullptr, *cell_w = nullptr, *cell_b = nullptr;
    int nwh = 0, nwm = 0, nwo = 0, nwa = 0, nwcl = 0, n640 = 0;
    for (int i = 0; i < n_in; ++i) {
        const float* p = (const float*)d_in[i];
        switch (in_sizes[i]) {
            case 1290240: forc = p; break;
            case 184320:  init = p; break;
            case 221184:  stat = p; break;
            case 1843200: tgt  = p; break;
            case 112000:  wx[0] = p; break;
            case 716800:  wx[1] = p; break;
            case 409600:  if (nwh  < 2) wh[nwh++]   = p; break;
            case 307200:  if (nwm  < 2) wm[nwm++]   = p; break;
            case 204800:  if (nwo  < 2) wo[nwo++]   = p; break;
            case 44800:   if (nwa  < 2) wa[nwa++]   = p; break;
            case 8192:    if (nwcl < 2) wcl[nwcl++] = p; break;
            case 4096:    adapter = p; break;
            case 1536:    cell_w  = p; break;
            case 128:     cell_b  = p; break;
            case 64:      mem_b   = p; break;
            case 640:     if (n640++ == 0) conv_last = p; else mem_w = p; break;
            default: break;
        }
    }
    if (!adapter && n_in >= 5) {  // params passed as one flat blob (pytree order)
        const float* q = (const float*)d_in[4];
        adapter = q; q += 4096;  cell_b = q; q += 128;  cell_w = q; q += 1536;
        for (int l = 0; l < 2; ++l) {
            wa[l] = q;  q += 44800;   wcl[l] = q; q += 8192;
            wh[l] = q;  q += 409600;  wm[l]  = q; q += 307200;
            wo[l] = q;  q += 204800;  wx[l]  = q; q += (l == 0 ? 112000 : 716800);
        }
        conv_last = q; q += 640;  mem_b = q; q += 64;  mem_w = q;
    }

    // ---- workspace layout (floats) ----
    float* ws = (float*)d_ws;
    size_t o = 0;
    float* net    = ws + o; o += (size_t)B_ * IMGC * HW_;       // 184320
    float* memory = ws + o; o += (size_t)B_ * NH_ * HW_;
    float* h0     = ws + o; o += (size_t)B_ * NH_ * HW_;
    float* h1     = ws + o; o += (size_t)B_ * NH_ * HW_;
    float* c0     = ws + o; o += (size_t)B_ * NH_ * HW_;
    float* c1     = ws + o; o += (size_t)B_ * NH_ * HW_;
    float* xc     = ws + o; o += (size_t)B_ * 7 * NH_ * HW_;    // 448 ch
    float* hconv  = ws + o; o += (size_t)B_ * 4 * NH_ * HW_;
    float* aconv  = ws + o; o += (size_t)B_ * 4 * NH_ * HW_;
    float* mconv  = ws + o; o += (size_t)B_ * 3 * NH_ * HW_;
    float* memcat = ws + o; o += (size_t)B_ * 2 * NH_ * HW_;    // also c_all
    float* oconv  = ws + o; o += (size_t)B_ * NH_ * HW_;        // also yc
    float* wclb   = ws + o; o += (size_t)B_ * NH_ * HW_;        // also ym
    float* dcb    = ws + o; o += (size_t)B_ * NH_ * HW_;
    float* dmb    = ws + o; o += (size_t)B_ * NH_ * HW_;
    float* accums = ws + o; o += 8;                              // [0]=sse [1]=dec

    const long bs_img = (long)IMGC * HW_;
    const long bs_nh  = (long)NH_ * HW_;
    const long bs_act = (long)T_ * ACTC * HW_;  // forcings batch stride

    // ---- init ----
    hipMemsetAsync(h0, 0, (size_t)B_ * NH_ * HW_ * sizeof(float), stream);
    hipMemsetAsync(h1, 0, (size_t)B_ * NH_ * HW_ * sizeof(float), stream);
    hipMemsetAsync(accums, 0, 8 * sizeof(float), stream);
    hipMemcpyAsync(net, init, (size_t)B_ * IMGC * HW_ * sizeof(float),
                   hipMemcpyDeviceToDevice, stream);

    launch_conv(init, bs_img, mem_w, mem_b, memory, IMGC, NH_, 1, stream);
    launch_conv(stat, (long)STATC * HW_, cell_w, cell_b, memcat, STATC, 2 * NH_, 1, stream);
    {
        long tot = (long)B_ * 2 * NH_ * HW_;
        split_c_kernel<<<(tot + 255) / 256, 256, 0, stream>>>(memcat, c0, c1);
    }

    const long totNH = (long)B_ * NH_ * HW_;
    const int  gNH   = (int)((totNH + 255) / 256);
    const float dec_scale = 1.0f / (float)(B_ * NH_ * T_ * 1 /* NUM_LAYERS-1 */);

    for (int t = 0; t < T_; ++t) {
        const float* a_t = forc + (size_t)t * ACTC * HW_;
        float* hs[2] = { h0, h1 };
        float* cs[2] = { c0, c1 };
        for (int l = 0; l < 2; ++l) {
            const float* xin = (l == 0) ? net : h0;
            long xbs = (l == 0) ? bs_img : bs_nh;
            int  xci = (l == 0) ? IMGC   : NH_;

            launch_conv(xin,  xbs,   wx[l], nullptr, xc,    xci, 7 * NH_, 5, stream);
            launch_conv(hs[l], bs_nh, wh[l], nullptr, hconv, NH_, 4 * NH_, 5, stream);
            launch_conv(a_t,  bs_act, wa[l], nullptr, aconv, ACTC, 4 * NH_, 5, stream);
            launch_conv(memory, bs_nh, wm[l], nullptr, mconv, NH_, 3 * NH_, 5, stream);

            cell_pointwise_kernel<<<gNH, 256, 0, stream>>>(
                xc, hconv, aconv, mconv, cs[l], memory, memcat, dcb, dmb);

            launch_conv(memcat, 2 * bs_nh, wo[l],  nullptr, oconv, 2 * NH_, NH_, 5, stream);
            launch_conv(memcat, 2 * bs_nh, wcl[l], nullptr, wclb,  2 * NH_, NH_, 1, stream);

            cell_output_kernel<<<gNH, 256, 0, stream>>>(xc, hconv, aconv, oconv, wclb, hs[l]);

            if (l >= 1) {  // decoupling loss for layers 1..L-1 (layer-0 result is discarded)
                launch_conv(dcb, bs_nh, adapter, nullptr, oconv, NH_, NH_, 1, stream);
                launch_conv(dmb, bs_nh, adapter, nullptr, wclb,  NH_, NH_, 1, stream);
                cos_reduce_kernel<<<B_ * NH_, 256, 0, stream>>>(oconv, wclb,
                                                                accums + 1, dec_scale);
            }
        }
        long totF = (long)B_ * IMGC * HW_;
        frame_kernel<<<(totF + 255) / 256, 256, 0, stream>>>(h1, conv_last, net,
                                                             (float*)d_out, t);
    }

    long n = (long)B_ * T_ * IMGC * HW_;
    mse_kernel<<<1024, 256, 0, stream>>>((const float*)d_out, tgt, accums, n);
    finalize_kernel<<<1, 1, 0, stream>>>(accums, (float*)d_out + n, n);
}
